// MultimodalMoE_33509334844055
// MI455X (gfx1250) — matrix operand order repaired and checked by reading, not compiled, not run on hardware
//
#include <hip/hip_runtime.h>
#include <hip/hip_bf16.h>
#include <math.h>

// ---------------------------------------------------------------------------
// MoE forward for gfx1250 (MI455X): bf16 WMMA expert FFN, fp32 in/out.
// TDM (tensor_load_to_lds) staging for activations, VALU cvt for fp32 weights.
// ---------------------------------------------------------------------------

typedef __bf16 bf16_t;
typedef __attribute__((ext_vector_type(16))) __bf16 v16bf;
typedef __attribute__((ext_vector_type(8)))  __bf16 v8bf;
typedef __attribute__((ext_vector_type(8)))  float  v8f;
typedef __attribute__((ext_vector_type(4)))  unsigned int u32x4;
typedef __attribute__((ext_vector_type(8)))  unsigned int u32x8;

constexpr int T_TOK = 8192;          // 4 * 2048 tokens
constexpr int D_DIM = 1024;
constexpr int F_DIM = 4096;
constexpr int E_EXP = 32;
constexpr int CAP   = 615;           // ceil(8192*2/32 * 1.2)
constexpr int CP    = 640;           // capacity padded to multiple of 64
constexpr int TK    = T_TOK * 2;     // top-2 slots
constexpr int SCAN_B = TK / 256;     // 64 scan blocks

// ---------------------------------------------------------------------------
// Wave32 reductions
// ---------------------------------------------------------------------------
__device__ inline float wave_max(float v) {
#pragma unroll
  for (int m = 16; m > 0; m >>= 1) v = fmaxf(v, __shfl_xor(v, m, 32));
  return v;
}
__device__ inline float wave_sum(float v) {
#pragma unroll
  for (int m = 16; m > 0; m >>= 1) v += __shfl_xor(v, m, 32);
  return v;
}

// ---------------------------------------------------------------------------
// Router: one wave per token, lane = expert. logits -> softmax -> top-2.
// ---------------------------------------------------------------------------
__global__ __launch_bounds__(256) void moe_router(
    const float* __restrict__ x, const float* __restrict__ gw,
    const float* __restrict__ gb, int* __restrict__ top_i,
    float* __restrict__ top_w)
{
  const int wave = threadIdx.x >> 5;
  const int lane = threadIdx.x & 31;
  const int t = blockIdx.x * 8 + wave;
  const float* xp = x + (size_t)t * D_DIM;

  float acc = gb[lane];
  for (int d = 0; d < D_DIM; d += 4) {
    const float4 xv = *(const float4*)(xp + d);
    acc = fmaf(xv.x, gw[(d + 0) * E_EXP + lane], acc);
    acc = fmaf(xv.y, gw[(d + 1) * E_EXP + lane], acc);
    acc = fmaf(xv.z, gw[(d + 2) * E_EXP + lane], acc);
    acc = fmaf(xv.w, gw[(d + 3) * E_EXP + lane], acc);
  }

  const float mx   = wave_max(acc);
  const float p    = __expf(acc - mx);
  const float s    = wave_sum(p);
  const float prob = p / s;

  const float m1 = wave_max(prob);
  const unsigned long long b1 = __ballot(prob == m1);
  const int i1 = __ffsll(b1) - 1;
  const float pm = (lane == i1) ? -1.0f : prob;
  const float m2 = wave_max(pm);
  const unsigned long long b2 = __ballot(pm == m2);
  const int i2 = __ffsll(b2) - 1;

  if (lane == 0) {
    const float inv = 1.0f / (m1 + m2);
    top_i[t * 2 + 0] = i1; top_w[t * 2 + 0] = m1 * inv;
    top_i[t * 2 + 1] = i2; top_w[t * 2 + 1] = m2 * inv;
  }
}

// ---------------------------------------------------------------------------
// Parallel token-major capacity scan (3 phases), exact reference semantics:
//   pos(j) = #{j' < j : expert(j') == expert(j)}  (then capacity-clipped)
// ---------------------------------------------------------------------------
__global__ __launch_bounds__(256) void moe_hist(
    const int* __restrict__ top_i, int* __restrict__ bhist)
{
  __shared__ int h[E_EXP];
  const int tid = threadIdx.x;
  if (tid < E_EXP) h[tid] = 0;
  __syncthreads();
  atomicAdd(&h[top_i[blockIdx.x * 256 + tid]], 1);
  __syncthreads();
  if (tid < E_EXP) bhist[blockIdx.x * E_EXP + tid] = h[tid];
}

__global__ void moe_prefix(const int* __restrict__ bhist,
                           int* __restrict__ pref, int* __restrict__ counts)
{
  const int e = threadIdx.x;  // 0..31
  int run = 0;
  for (int b = 0; b < SCAN_B; ++b) {
    pref[b * E_EXP + e] = run;
    run += bhist[b * E_EXP + e];
  }
  counts[e] = run < CAP ? run : CAP;
}

__global__ __launch_bounds__(256) void moe_rank(
    const int* __restrict__ top_i, const int* __restrict__ pref,
    int* __restrict__ pos, int* __restrict__ slot)
{
  __shared__ int le[256];
  const int tid = threadIdx.x;
  const int j = blockIdx.x * 256 + tid;
  const int e = top_i[j];
  le[tid] = e;
  __syncthreads();
  int rank = 0;
  for (int i = 0; i < tid; ++i) rank += (le[i] == e) ? 1 : 0;
  const int p = pref[blockIdx.x * E_EXP + e] + rank;
  if (p < CAP) { pos[j] = p; slot[e * CP + p] = j; }
  else         { pos[j] = -1; }
}

// ---------------------------------------------------------------------------
// Scatter tokens into xe[E][CP][D] as bf16; zero-fill unused capacity rows.
// ---------------------------------------------------------------------------
__global__ __launch_bounds__(128) void moe_scatter(
    const float* __restrict__ x, const int* __restrict__ slot,
    const int* __restrict__ counts, bf16_t* __restrict__ xe)
{
  const int e = blockIdx.y;
  const int c = blockIdx.x;
  bf16_t* dst = xe + ((size_t)e * CP + c) * D_DIM;
  const int i = threadIdx.x * 8;  // 128 threads * 8 = 1024 = D
  if (c < counts[e]) {
    const int j = slot[e * CP + c];
    const float* src = x + (size_t)(j >> 1) * D_DIM;
    const float4 f0 = *(const float4*)(src + i);
    const float4 f1 = *(const float4*)(src + i + 4);
    v8bf o;
    o[0] = (bf16_t)f0.x; o[1] = (bf16_t)f0.y; o[2] = (bf16_t)f0.z; o[3] = (bf16_t)f0.w;
    o[4] = (bf16_t)f1.x; o[5] = (bf16_t)f1.y; o[6] = (bf16_t)f1.z; o[7] = (bf16_t)f1.w;
    *(v8bf*)(dst + i) = o;
  } else {
    v8bf z = {};
    *(v8bf*)(dst + i) = z;
  }
}

// ---------------------------------------------------------------------------
// Grouped GEMM: Out[e, m, n] = act( A[e, m, :] . B[e, :, n] + bias[e, n] )
//   A : bf16 [E][CP][K]  -> staged by the Tensor Data Mover (TENSORcnt),
//                           one 64x32 strided-tile descriptor per K-step,
//                           D# padding deposits rows at LDS pitch AP=40.
//   B : f32  [E][K][N]   -> converted to bf16 + transposed to [N][K] in LDS
// Block tile 64(M) x 256(N), 8 waves (2M x 4N), wave tile 32x64
//   -> 8 v_wmma_f32_16x16x32_bf16 per K-step per wave.
// ---------------------------------------------------------------------------
template <bool GELU, typename OutT>
__global__ __launch_bounds__(256) void moe_gemm(
    const bf16_t* __restrict__ A, const float* __restrict__ B,
    const float* __restrict__ bias, OutT* __restrict__ Out,
    const int Kk, const int Nn)
{
  constexpr int AP = 40;  // LDS row pitch (bf16): 80B, 16B-aligned rows
  __shared__ __align__(16) bf16_t lA[64 * AP];    // [m][k]
  __shared__ __align__(16) bf16_t lB[256 * AP];   // [n][k] (transposed weight)

  const int e  = blockIdx.z;
  const int m0 = blockIdx.y * 64;
  const int n0 = blockIdx.x * 256;

  const int tid  = threadIdx.x;
  const int lane = tid & 31;
  const int wv   = tid >> 5;
  const int wn   = wv & 3;    // 4 N-groups of 64
  const int wmw  = wv >> 2;   // 2 M-groups of 32
  const int l16  = lane & 15;
  const int half = lane >> 4;

  const bf16_t* Ag = A + (size_t)e * CP * Kk + (size_t)m0 * Kk;
  const float*  Bg = B + (size_t)e * Kk * Nn;

  v8f acc[2][4] = {};

  const int brow = tid >> 5;        // 0..7
  const int bcol = tid & 31;        // 0..31
  // Low 32 bits of a flat shared pointer are the LDS byte address.
  const unsigned int ldsA = (unsigned int)(size_t)&lA[0];
  const unsigned long long gA0 = (unsigned long long)(size_t)Ag;

  for (int kt = 0; kt < Kk; kt += 32) {
    // --- A tile (64 rows x 64B, row stride Kk*2B): one TDM descriptor ---
    if (wv == 0) {
      const unsigned long long ga = gA0 + (unsigned long long)kt * 2ull;
      u32x4 g0;
      g0[0] = 1u;                                      // count=1, user mode
      g0[1] = ldsA;                                    // lds_addr
      g0[2] = (unsigned int)ga;                        // global_addr[31:0]
      g0[3] = (unsigned int)(ga >> 32) | (2u << 30);   // addr[56:32] | type=2
      u32x8 g1;
      g1[0] = 0x06D10000u;  // data_size=2B | pad_en | interval=16dw | pad=4dw
      g1[1] = ((unsigned int)Kk & 0xFFFFu) << 16;      // tensor_dim0 lo
      g1[2] = ((unsigned int)Kk >> 16) | (64u << 16);  // dim0 hi | tensor_dim1=64
      g1[3] = 32u << 16;                               // tile_dim0 = 32
      g1[4] = 64u;                                     // tile_dim1 = 64
      g1[5] = (unsigned int)Kk;                        // tensor_dim0_stride
      g1[6] = 0u;
      g1[7] = 0u;
      asm volatile("tensor_load_to_lds %0, %1" :: "s"(g0), "s"(g1) : "memory");
    }
    // --- B tile (32x256 f32): coalesced read, cvt to bf16, transpose ---
#pragma unroll
    for (int i = 0; i < 4; ++i) {
      const int r = brow + i * 8;  // k within tile, 0..31
      const float* browp = Bg + (size_t)(kt + r) * Nn + n0;
#pragma unroll
      for (int j = 0; j < 8; ++j) {
        const int n = bcol + 32 * j;
        lB[n * AP + r] = (bf16_t)browp[n];
      }
    }
    if (kt + 32 < Kk)  // hint next weight tile toward L2
      __builtin_prefetch(Bg + (size_t)(kt + 32 + brow) * Nn + n0 + bcol * 4, 0, 0);

    if (wv == 0) {
#if __has_builtin(__builtin_amdgcn_s_wait_tensorcnt)
      __builtin_amdgcn_s_wait_tensorcnt(0);
#else
      asm volatile("s_wait_tensorcnt 0x0" ::: "memory");
#endif
    }
    __syncthreads();

    // A fragments: lane<16 -> K {0..7,16..23}; lane>=16 -> K {8..15,24..31}
    v16bf af[2];
#pragma unroll
    for (int fi = 0; fi < 2; ++fi) {
      const bf16_t* ap = &lA[(wmw * 32 + fi * 16 + l16) * AP + (half ? 8 : 0)];
      const v8bf a0 = *(const v8bf*)ap;
      const v8bf a1 = *(const v8bf*)(ap + 16);
      af[fi] = __builtin_shufflevector(
          a0, a1, 0, 1, 2, 3, 4, 5, 6, 7, 8, 9, 10, 11, 12, 13, 14, 15);
    }
    // B fragments: lane<16 -> K {0..7,16..23}; lane>=16 -> K {8..15,24..31} (reused across fi)
#pragma unroll
    for (int i = 0; i < 4; ++i) {
      const bf16_t* bp = &lB[(wn * 64 + i * 16 + l16) * AP + (half ? 8 : 0)];
      const v8bf b0  = *(const v8bf*)bp;
      const v8bf b1v = *(const v8bf*)(bp + 16);
      const v16bf bfr = __builtin_shufflevector(
          b0, b1v, 0, 1, 2, 3, 4, 5, 6, 7, 8, 9, 10, 11, 12, 13, 14, 15);
      acc[0][i] = __builtin_amdgcn_wmma_f32_16x16x32_bf16(
          false, af[0], false, bfr, (short)0, acc[0][i], false, false);
      acc[1][i] = __builtin_amdgcn_wmma_f32_16x16x32_bf16(
          false, af[1], false, bfr, (short)0, acc[1][i], false, false);
    }
    __syncthreads();
  }

  // Epilogue: D layout -> VGPR r: row = r + 8*half, col = l16 (coalesced N)
  const int nb = n0 + wn * 64 + l16;
#pragma unroll
  for (int fi = 0; fi < 2; ++fi) {
    const int mb = m0 + wmw * 32 + fi * 16 + half * 8;
#pragma unroll
    for (int i = 0; i < 4; ++i) {
      const int n = nb + i * 16;
      const float bb = bias[(size_t)e * Nn + n];
#pragma unroll
      for (int r = 0; r < 8; ++r) {
        float v = acc[fi][i][r] + bb;
        if (GELU) v = 0.5f * v * (1.0f + erff(v * 0.70710678f));  // exact gelu
        Out[((size_t)(e * CP) + mb + r) * Nn + n] = (OutT)v;
      }
    }
  }
}

// ---------------------------------------------------------------------------
// Combine: y[t] = sum_k gate_w[t,k] * valid * ye[e_k, pos_k]
// ---------------------------------------------------------------------------
__global__ __launch_bounds__(256) void moe_combine(
    const float* __restrict__ ye, const int* __restrict__ top_i,
    const float* __restrict__ top_w, const int* __restrict__ pos,
    float* __restrict__ y)
{
  const int t = blockIdx.x;
  const int d = threadIdx.x * 4;  // 256 * 4 = 1024 = D
  float4 acc = {0.f, 0.f, 0.f, 0.f};
#pragma unroll
  for (int k = 0; k < 2; ++k) {
    const int j = t * 2 + k;
    const int p = pos[j];
    if (p >= 0) {
      const int   e = top_i[j];
      const float w = top_w[j];
      const float4 v = *(const float4*)(ye + ((size_t)e * CP + p) * D_DIM + d);
      acc.x += w * v.x; acc.y += w * v.y; acc.z += w * v.z; acc.w += w * v.w;
    }
  }
  *(float4*)(y + (size_t)t * D_DIM + d) = acc;
}

// ---------------------------------------------------------------------------
// Launch
// ---------------------------------------------------------------------------
extern "C" void kernel_launch(void* const* d_in, const int* in_sizes, int n_in,
                              void* d_out, int out_size, void* d_ws, size_t ws_size,
                              hipStream_t stream)
{
  const float* x  = (const float*)d_in[0];
  const float* gw = (const float*)d_in[1];
  const float* gb = (const float*)d_in[2];
  const float* w1 = (const float*)d_in[3];
  const float* b1 = (const float*)d_in[4];
  const float* w2 = (const float*)d_in[5];
  const float* b2 = (const float*)d_in[6];
  float* y = (float*)d_out;

  char* ws = (char*)d_ws;
  size_t off = 0;
  auto take = [&](size_t bytes) -> char* {
    char* p = ws + off;
    off = (off + bytes + 255) & ~(size_t)255;
    return p;
  };
  bf16_t* xe    = (bf16_t*)take((size_t)E_EXP * CP * D_DIM * sizeof(bf16_t));
  bf16_t* he    = (bf16_t*)take((size_t)E_EXP * CP * F_DIM * sizeof(bf16_t));
  float*  ye    = (float*) take((size_t)E_EXP * CP * D_DIM * sizeof(float));
  int*    top_i = (int*)   take((size_t)TK * sizeof(int));
  float*  top_w = (float*) take((size_t)TK * sizeof(float));
  int*    pos   = (int*)   take((size_t)TK * sizeof(int));
  int*    slot  = (int*)   take((size_t)E_EXP * CP * sizeof(int));
  int*    cnts  = (int*)   take((size_t)E_EXP * sizeof(int));
  int*    bhist = (int*)   take((size_t)SCAN_B * E_EXP * sizeof(int));
  int*    pref  = (int*)   take((size_t)SCAN_B * E_EXP * sizeof(int));

  moe_router <<<T_TOK / 8, 256, 0, stream>>>(x, gw, gb, top_i, top_w);
  moe_hist   <<<SCAN_B, 256, 0, stream>>>(top_i, bhist);
  moe_prefix <<<1, 32, 0, stream>>>(bhist, pref, cnts);
  moe_rank   <<<SCAN_B, 256, 0, stream>>>(top_i, pref, pos, slot);
  moe_scatter<<<dim3(CP, E_EXP), 128, 0, stream>>>(x, slot, cnts, xe);
  moe_gemm<true,  bf16_t><<<dim3(F_DIM / 256, CP / 64, E_EXP), 256, 0, stream>>>(
      xe, w1, b1, he, D_DIM, F_DIM);
  moe_gemm<false, float ><<<dim3(D_DIM / 256, CP / 64, E_EXP), 256, 0, stream>>>(
      he, w2, b2, ye, F_DIM, D_DIM);
  moe_combine<<<T_TOK, 256, 0, stream>>>(ye, top_i, top_w, pos, y);
}
